// SelfNonParametric_70531952935514
// MI455X (gfx1250) — matrix operand order repaired and checked by reading, not compiled, not run on hardware
//
#include <hip/hip_runtime.h>
#include <cmath>

typedef __attribute__((ext_vector_type(16))) _Float16 v16h;
typedef __attribute__((ext_vector_type(8)))  _Float16 v8h;
typedef __attribute__((ext_vector_type(8)))  float    v8f;
typedef __attribute__((ext_vector_type(4)))  int      v4i;

#define N_LB   2048
#define N_ULB  7168
#define NTOT   16384      // 2048 + 2*7168
#define D      128
#define C      100
#define CP     112        // C padded to 7 tiles of 16
#define SQRT10 3.16227766016838f
#define SMAX   10.0f      // |S| <= 10 by construction (cos * 1/tau), softmax shift

// d_out layout (floats): anchor(917504) positive(917504) lb_feat(262144)
//                        lb_one_hot(204800) out_lb(204800) ulb_a(716800) ulb_b(716800)
#define OFF_OUTLB 2301952u
#define OFF_ULB   2506752u
#define N_PASS    2301952u

#if defined(__has_builtin)
#  if __has_builtin(__builtin_amdgcn_global_load_async_to_lds_b128)
#    define HAVE_ASYNC_LDS 1
#  endif
#endif
#ifndef HAVE_ASYNC_LDS
#  define HAVE_ASYNC_LDS 0
#endif

typedef __attribute__((address_space(1))) v4i gv4i;
typedef __attribute__((address_space(3))) v4i lv4i;

// Stage one 16-byte chunk global -> LDS (async DMA path on CDNA5, fallback copy).
__device__ __forceinline__ void stage16(const _Float16* g, _Float16* l) {
#if HAVE_ASYNC_LDS
    __builtin_amdgcn_global_load_async_to_lds_b128((gv4i*)g, (lv4i*)l, 0, 0);
#else
    *(v8h*)l = *(const v8h*)g;
#endif
}

__device__ __forceinline__ void wait_stage() {
#if HAVE_ASYNC_LDS
#  if __has_builtin(__builtin_amdgcn_s_wait_asynccnt)
    __builtin_amdgcn_s_wait_asynccnt(0);
#  else
    asm volatile("s_wait_asynccnt 0x0" ::: "memory");
#  endif
#endif
}

// 2-bit bit-reverse of the 16B part index within a 32-half k-chunk: 0,2,1,3.
// Storing A-operand tiles with sub-blocks [8..15]<->[16..23] swapped makes the
// WMMA A-fragment a single contiguous 32B load (k is a reduction dim, so any
// consistent permutation is legal).
__device__ __forceinline__ int swapq(int q) { return ((q & 1) << 1) | (q >> 1); }

// ---------------- prep 1: normalized, sqrt(10)-scaled f16 features ----------------
__global__ __launch_bounds__(128) void prep_feats(const float* __restrict__ lbf,
                                                  const float* __restrict__ anc,
                                                  const float* __restrict__ pos,
                                                  _Float16* __restrict__ F) {
    int lane = threadIdx.x & 31;
    int wave = threadIdx.x >> 5;
    int row  = blockIdx.x * 4 + wave;
    if (row >= NTOT) return;
    const float* src = (row < N_LB) ? (lbf + (size_t)row * D)
                     : (row < N_LB + N_ULB) ? (anc + (size_t)(row - N_LB) * D)
                                            : (pos + (size_t)(row - N_LB - N_ULB) * D);
    float x[4]; float ss = 0.f;
#pragma unroll
    for (int i = 0; i < 4; ++i) { x[i] = src[lane + 32 * i]; ss += x[i] * x[i]; }
#pragma unroll
    for (int m = 1; m < 32; m <<= 1) ss += __shfl_xor(ss, m, 32);
    float sc = SQRT10 / fmaxf(sqrtf(ss), 1e-12f);
#pragma unroll
    for (int i = 0; i < 4; ++i)
        F[(size_t)row * D + lane + 32 * i] = (_Float16)(x[i] * sc);
}

// ---------------- prep 2: class_val (one-hot | softmax(logits)) transposed f16 -----
__global__ __launch_bounds__(128) void prep_classval(const float* __restrict__ lb1h,
                                                     const float* __restrict__ lg1,
                                                     const float* __restrict__ lg2,
                                                     _Float16* __restrict__ VT) {
    int lane = threadIdx.x & 31;
    int wave = threadIdx.x >> 5;
    int row  = blockIdx.x * 4 + wave;
    if (row >= NTOT) return;
    if (row < N_LB) {
        const float* src = lb1h + (size_t)row * C;
#pragma unroll
        for (int i = 0; i < 4; ++i) {
            int c = lane + 32 * i;
            if (c < CP)
                VT[(size_t)c * NTOT + row] = (c < C) ? (_Float16)src[c] : (_Float16)0.f;
        }
        return;
    }
    int ur = row - N_LB;
    const float* lg = (ur < N_ULB) ? (lg1 + (size_t)ur * C)
                                   : (lg2 + (size_t)(ur - N_ULB) * C);
    float x[4];
#pragma unroll
    for (int i = 0; i < 4; ++i) {
        int c = lane + 32 * i;
        x[i] = (c < C) ? lg[c] : -__builtin_inff();
    }
    float mx = fmaxf(fmaxf(x[0], x[1]), fmaxf(x[2], x[3]));
#pragma unroll
    for (int m = 1; m < 32; m <<= 1) mx = fmaxf(mx, __shfl_xor(mx, m, 32));
    float e[4]; float sum = 0.f;
#pragma unroll
    for (int i = 0; i < 4; ++i) {
        int c = lane + 32 * i;
        e[i] = (c < C) ? __expf(x[i] - mx) : 0.f;
        sum += e[i];
    }
#pragma unroll
    for (int m = 1; m < 32; m <<= 1) sum += __shfl_xor(sum, m, 32);
    float rs = 1.f / sum;
#pragma unroll
    for (int i = 0; i < 4; ++i) {
        int c = lane + 32 * i;
        if (c < CP) VT[(size_t)c * NTOT + row] = (_Float16)(e[i] * rs);
    }
}

// ---------------- pass-through copies ---------------------------------------------
__global__ __launch_bounds__(256) void copy_pass(const float* __restrict__ anc,
                                                 const float* __restrict__ pos,
                                                 const float* __restrict__ lbf,
                                                 const float* __restrict__ lb1h,
                                                 float* __restrict__ out) {
    for (unsigned i = blockIdx.x * 256u + threadIdx.x; i < N_PASS; i += gridDim.x * 256u) {
        float v;
        if      (i <  917504u) v = anc[i];
        else if (i < 1835008u) v = pos[i -  917504u];
        else if (i < 2097152u) v = lbf[i - 1835008u];
        else                   v = lb1h[i - 2097152u];
        out[i] = v;
    }
}

// ---------------- flash attention: softmax(F·Fᵀ) · V, never materialized ----------
// Transposed formulation: S^T = K·Q^T and O^T = V^T·P^T; the exponentiated tile
// converts C/D layout -> next GEMM's B layout with one half-wave shuffle.
// Fixed-shift softmax (|S|<=10). 4 waves/block share double-buffered K/V tiles
// staged via async global->LDS DMA in global row order; using k = kc*32 + hi*16 + s so
// every WMMA fragment is one contiguous 32B load.
__global__ __launch_bounds__(128) void flash_attn(const _Float16* __restrict__ F,
                                                  const _Float16* __restrict__ VT,
                                                  float* __restrict__ out) {
    __shared__ __align__(32) _Float16 Kb[2][32 * D];    // 2 x 8 KB, row-major layout
    __shared__ __align__(32) _Float16 Vb[2][CP * 32];   // 2 x 7 KB, row-major layout

    const int tid   = threadIdx.x;
    const int lane  = tid & 31;
    const int wave  = tid >> 5;
    const int nlo   = lane & 15;
    const int hi    = lane >> 4;
    const int qbase = (blockIdx.x * 4 + wave) * 16;

    // Cooperative staging of one 32-row K tile (512 x 16B) and V slice (448 x 16B);
    // the LDS destination keeps the 16B parts in global order.
    auto load_tiles = [&](int j, int buf) {
#pragma unroll
        for (int i = 0; i < 4; ++i) {
            int id    = tid + 128 * i;            // 0..511
            int row   = id >> 4, part = id & 15;
            int dpart = part;
            stage16(F + (size_t)(j + row) * D + part * 8, &Kb[buf][row * D + dpart * 8]);
        }
#pragma unroll
        for (int i = 0; i < 4; ++i) {
            int id = tid + 128 * i;               // 0..511, use 0..447
            if (id < CP * 4) {
                int c = id >> 2, part = id & 3;
                stage16(VT + (size_t)c * NTOT + j + part * 8,
                        &Vb[buf][c * 32 + part * 8]);
            }
        }
    };

    // Q as 4 B-fragments (32x16 f16): lane col m=nlo, halves k = kc*32 + hi*16 + {0..15}
    v16h qb[4];
#pragma unroll
    for (int kc = 0; kc < 4; ++kc)
        qb[kc] = *(const v16h*)(F + (size_t)(qbase + nlo) * D + kc * 32 + hi * 16);

    v8f acc[7];                                    // O^T tiles: col m=nlo, row c=t*16+hi*8+r
#pragma unroll
    for (int t = 0; t < 7; ++t) acc[t] = (v8f){0.f, 0.f, 0.f, 0.f, 0.f, 0.f, 0.f, 0.f};
    float lsum = 0.f;                              // partial denominator for Q-row nlo

    load_tiles(0, 0);
    wait_stage();
    __syncthreads();

    int buf = 0;
    for (int j = 0; j < NTOT; j += 32) {
        if (j + 32 < NTOT) {
            load_tiles(j + 32, buf ^ 1);
            __builtin_prefetch(F + (size_t)(j + 64 + lane) * D, 0, 1);  // global_prefetch_b8
        }

        // K A-fragments (contiguous k = kc*32 + hi*16 + {0..15}, same order as qb).
        const _Float16* ka0 = &Kb[buf][nlo * D + hi * 16];
        const _Float16* ka1 = &Kb[buf][(16 + nlo) * D + hi * 16];
        v16h a0[4], a1[4];
#pragma unroll
        for (int kc = 0; kc < 4; ++kc) {
            a0[kc] = *(const v16h*)(ka0 + kc * 32);
            a1[kc] = *(const v16h*)(ka1 + kc * 32);
        }

        // S^T = K x Q^T (rows = K-rows, cols = Q-rows)
        v8f s0 = (v8f){0.f, 0.f, 0.f, 0.f, 0.f, 0.f, 0.f, 0.f};
        v8f s1 = s0;
#pragma unroll
        for (int kc = 0; kc < 4; ++kc) {
            s0 = __builtin_amdgcn_wmma_f32_16x16x32_f16(false, a0[kc], false, qb[kc],
                                                        (short)0, s0, false, false);
            s1 = __builtin_amdgcn_wmma_f32_16x16x32_f16(false, a1[kc], false, qb[kc],
                                                        (short)0, s1, false, false);
        }

        // Fixed-shift numerators; lane's 16 e-values all belong to Q-row nlo.
        float e0[8], e1[8];
#pragma unroll
        for (int r = 0; r < 8; ++r) {
            e0[r] = __expf(s0[r] - SMAX);
            e1[r] = __expf(s1[r] - SMAX);
            lsum += e0[r] + e1[r];
        }

        // Build P^T B-fragment (col m=nlo, halves k = hi*16 + {0..15}) with one
        // half-wave exchange instead of an LDS transpose round-trip.
        v16h pb;
#pragma unroll
        for (int r = 0; r < 8; ++r) {
            float o0 = __shfl_xor(e0[r], 16, 32);
            float o1 = __shfl_xor(e1[r], 16, 32);
            pb[r]     = (_Float16)(hi ? o1 : e0[r]);   // k = hi*16 + r
            pb[8 + r] = (_Float16)(hi ? e1[r] : o0);   // k = hi*16 + 8 + r
        }

        // O^T += V^T(112x32) x P^T(32x16): 7 A-tiles, each one contiguous 32B load.
#pragma unroll
        for (int t = 0; t < 7; ++t) {
            v16h va = *(const v16h*)(&Vb[buf][(t * 16 + nlo) * 32 + hi * 16]);
            acc[t] = __builtin_amdgcn_wmma_f32_16x16x32_f16(false, va, false, pb,
                                                            (short)0, acc[t], false, false);
        }

        wait_stage();        // my async copies for buf^1 have landed
        __syncthreads();     // all waves done with buf; next tiles visible
        buf ^= 1;
    }

    // Denominator for Q-row nlo: combine the two half-groups once.
    float tot = lsum + __shfl_xor(lsum, 16, 32);
    float inv = 1.f / tot;

    int g = qbase + nlo;
    float* orow = (g < N_LB) ? (out + OFF_OUTLB + (size_t)g * C)
                             : (out + OFF_ULB + (size_t)(g - N_LB) * C);
#pragma unroll
    for (int t = 0; t < 7; ++t)
#pragma unroll
        for (int r = 0; r < 8; ++r) {
            int c = t * 16 + hi * 8 + r;
            if (c < C) orow[c] = acc[t][r] * inv;
        }
}

extern "C" void kernel_launch(void* const* d_in, const int* in_sizes, int n_in,
                              void* d_out, int out_size, void* d_ws, size_t ws_size,
                              hipStream_t stream) {
    const float* anchor = (const float*)d_in[0];
    const float* positi = (const float*)d_in[1];
    const float* lbfeat = (const float*)d_in[2];
    const float* lb1hot = (const float*)d_in[3];
    // d_in[4] = logits_x_lb: only its shape matters in the reference.
    const float* lulb1  = (const float*)d_in[5];
    const float* lulb2  = (const float*)d_in[6];
    float* out = (float*)d_out;

    _Float16* F  = (_Float16*)d_ws;                       // 16384*128 f16 = 4 MB
    _Float16* VT = F + (size_t)NTOT * D;                  // 112*16384 f16 = 3.5 MB

    prep_feats   <<<NTOT / 4, 128, 0, stream>>>(lbfeat, anchor, positi, F);
    prep_classval<<<NTOT / 4, 128, 0, stream>>>(lb1hot, lulb1, lulb2, VT);
    copy_pass    <<<2048, 256, 0, stream>>>(anchor, positi, lbfeat, lb1hot, out);
    flash_attn   <<<NTOT / 64, 128, 0, stream>>>(F, VT, out);
}
